// A2Conv_64630667870462
// MI455X (gfx1250) — matrix operand order repaired and checked by reading, not compiled, not run on hardware
//
#include <hip/hip_runtime.h>
#include <hip/hip_bf16.h>

typedef __attribute__((ext_vector_type(16))) _Float16 v16h;
typedef __attribute__((ext_vector_type(8)))  _Float16 v8h;
typedef __attribute__((ext_vector_type(8)))  float    v8f;

#define EMBED 128
#define KCAP  8
#define ASIZE 16
#define NEIB  16
#define WPB   8   // waves (nodes) per block

__device__ __forceinline__ void lds_fence() {
  // DS pipe is in-order within a wave; wait + memory clobber stops reordering
  asm volatile("s_wait_dscnt 0" ::: "memory");
}

// ---------------------------------------------------------------------------
// Kernel 1: per-capsule L2 normalize; emit f32 copy (self term) + f16 copy
// (gather table, halves L2 gather traffic; 25.6MB -> fully L2 resident).
// One thread per element; 16-lane shfl reduction = one capsule.
// ---------------------------------------------------------------------------
__global__ void a2_normalize(const float* __restrict__ feat,
                             float* h32, _Float16* __restrict__ h16, int total) {
  int t = blockIdx.x * blockDim.x + threadIdx.x;
  if (t >= total) return;
  float x  = feat[t];
  float ss = x * x;
  #pragma unroll
  for (int m = 1; m < 16; m <<= 1) ss += __shfl_xor(ss, m, 32);
  float hx = x / fmaxf(sqrtf(ss), 1e-12f);
  h32[t] = hx;
  h16[t] = (_Float16)hx;
}

// ---------------------------------------------------------------------------
// Kernel 2: one wave32 per node. Gather 16 neighbor rows (f16) into LDS in
// two layouts (A-layout [m][j] and B-layout [j][m]), then 3 routing rounds,
// each phase = 4 chained v_wmma_f32_16x16x32_f16.
// ---------------------------------------------------------------------------
__global__ __launch_bounds__(256, 1)
void a2_route(const float* hself,                 // normalized f32 (== out)
              const _Float16* __restrict__ h16,   // normalized f16 table
              const int* __restrict__ neighbors,
              const int* __restrict__ niter_ptr,
              float* out, int num_node) {
  __shared__ __align__(32) _Float16 s_zbuf[WPB][NEIB][EMBED];  // [m][k*16+a]
  __shared__ __align__(32) _Float16 s_zt  [WPB][EMBED][NEIB];  // [k*16+a][m]
  __shared__ __align__(32) _Float16 s_ubuf[WPB][KCAP][ASIZE];  // [k][a]

  const int  wave = threadIdx.x >> 5;
  const int  L    = threadIdx.x & 31;
  const int  l    = L & 15;
  const bool hi   = (L >= 16);
  const int  node = blockIdx.x * WPB + wave;
  if (node >= num_node) return;   // wave-uniform

  _Float16* zb = &s_zbuf[wave][0][0];
  _Float16* zt = &s_zt  [wave][0][0];
  _Float16* ub = &s_ubuf[wave][0][0];

  // ---- gather: lane pair (2m,2m+1) splits neighbor row m (256B) ----
  {
    const int m  = L >> 1;
    const int j0 = (L & 1) * 64;
    int  nbr   = neighbors[node * NEIB + m];
    bool valid = ((unsigned)nbr < (unsigned)num_node);  // pad row -> zeros
    const _Float16* src = h16 + (size_t)(valid ? nbr : 0) * EMBED + j0;
    #pragma unroll
    for (int t = 0; t < 8; ++t) {
      v8h v = *(const v8h*)(src + t * 8);
      if (!valid) {
        #pragma unroll
        for (int e = 0; e < 8; ++e) v[e] = (_Float16)0;
      }
      *(v8h*)(zb + m * EMBED + j0 + t * 8) = v;       // A layout
      #pragma unroll
      for (int e = 0; e < 8; ++e)                      // B layout (transpose)
        zt[(j0 + t * 8 + e) * NEIB + m] = v[e];
    }
  }
  lds_fence();

  // ---- self term preloaded in C/D layout: lane a=l holds u[k=0..7][a] ----
  v8f hacc;
  #pragma unroll
  for (int r = 0; r < 8; ++r) {
    float hv = hself[(size_t)node * EMBED + r * ASIZE + l];
    hacc[r] = hi ? 0.0f : hv;
  }

  int niter = niter_ptr[0];
  if (niter < 1)  niter = 1;
  if (niter > 16) niter = 16;

  _Float16 myp[8];
  v8f acc;

  for (int it = 0; it < niter; ++it) {
    if (it == 0) {
      #pragma unroll
      for (int r = 0; r < 8; ++r) myp[r] = (_Float16)0.125f;  // softmax(0)
    } else {
      // ===== dot phase: W[m,k] = sum_a z[m,k,a] * u[k,a] =====
      v16h uvec;  // u[k=l&7][0..15] (meaningful where selected below)
      {
        const _Float16* up = ub + (l & 7) * ASIZE;
        v8h u0 = *(const v8h*)(up);
        v8h u1 = *(const v8h*)(up + 8);
        #pragma unroll
        for (int e = 0; e < 8; ++e) { uvec[e] = u0[e]; uvec[e + 8] = u1[e]; }
      }
      v8f w;
      #pragma unroll
      for (int r = 0; r < 8; ++r) w[r] = 0.0f;
      #pragma unroll
      for (int c = 0; c < 4; ++c) {
        // A: z rows, contraction chunk j in [32c,32c+32)
        v16h A;
        {
          const _Float16* zr = zb + l * EMBED + 32 * c + (hi ? 16 : 0);
          v8h a0 = *(const v8h*)(zr);
          v8h a1 = *(const v8h*)(zr + 8);
          #pragma unroll
          for (int e = 0; e < 8; ++e) { A[e] = a0[e]; A[e + 8] = a1[e]; }
        }
        // B: block-diagonal expansion of u (one carrier lane per half-wave)
        bool sel = (!hi && l == 2 * c) || (hi && l == 2 * c + 1);
        v16h B;
        #pragma unroll
        for (int e = 0; e < 16; ++e) B[e] = sel ? uvec[e] : (_Float16)0;
        w = __builtin_amdgcn_wmma_f32_16x16x32_f16(false, A, false, B,
                                                   (short)0, w, false, false);
      }
      // ===== softmax over k: column k sits on lane k (rows m in VGPRs) =====
      #pragma unroll
      for (int r = 0; r < 8; ++r) {
        float v  = w[r];
        float mx = v;
        #pragma unroll
        for (int m = 1; m < 8; m <<= 1) mx = fmaxf(mx, __shfl_xor(mx, m, 32));
        float ex  = __expf(v - mx);
        float sum = ex;
        #pragma unroll
        for (int m = 1; m < 8; m <<= 1) sum += __shfl_xor(sum, m, 32);
        myp[r] = (_Float16)(ex / sum);   // p[m=r(+8 if hi)][k=l] stays put
      }
    }

    // ===== sum phase: u[k,a] = sum_m p[m,k] z[m,k,a] + h[k,a] =====
    acc = hacc;
    #pragma unroll
    for (int c = 0; c < 4; ++c) {
      // A2: block-diagonal expansion of p; lane already owns its p column
      v16h PA;
      #pragma unroll
      for (int e = 0; e < 8; ++e) {
        PA[e]     = (l == 2 * c)     ? myp[e] : (_Float16)0;
        PA[e + 8] = (l == 2 * c + 1) ? myp[e] : (_Float16)0;
      }
      // B2: z in [(k,m) x a] layout, 8 m (8h..8h+7) from capsule 2c then 2c+1
      v16h BZ;
      {
        const _Float16* zr = zt + ((2 * c) * ASIZE + l) * NEIB + (hi ? 8 : 0);
        v8h b0 = *(const v8h*)(zr);
        v8h b1 = *(const v8h*)(zr + ASIZE * NEIB);
        #pragma unroll
        for (int e = 0; e < 8; ++e) { BZ[e] = b0[e]; BZ[e + 8] = b1[e]; }
      }
      acc = __builtin_amdgcn_wmma_f32_16x16x32_f16(false, PA, false, BZ,
                                                   (short)0, acc, false, false);
    }

    if (it < niter - 1) {
      // per-capsule L2 norm over a (16 lanes)
      #pragma unroll
      for (int r = 0; r < 8; ++r) {
        float v  = acc[r];
        float ss = v * v;
        #pragma unroll
        for (int m = 1; m < 16; m <<= 1) ss += __shfl_xor(ss, m, 32);
        acc[r] = v / fmaxf(sqrtf(ss), 1e-12f);
      }
      if (!hi) {
        #pragma unroll
        for (int r = 0; r < 8; ++r) ub[r * ASIZE + l] = (_Float16)acc[r];
      }
      lds_fence();
    }
  }

  if (!hi) {
    #pragma unroll
    for (int r = 0; r < 8; ++r)
      out[(size_t)node * EMBED + r * ASIZE + l] = acc[r];
  }
}

// ---------------------------------------------------------------------------
extern "C" void kernel_launch(void* const* d_in, const int* in_sizes, int n_in,
                              void* d_out, int out_size, void* d_ws, size_t ws_size,
                              hipStream_t stream) {
  const float* feat      = (const float*)d_in[0];
  const int*   neighbors = (const int*)d_in[1];
  const int*   niter     = (const int*)d_in[2];
  float*       out       = (float*)d_out;

  int total    = in_sizes[0];        // num_node * 128
  int num_node = total / EMBED;
  _Float16* h16 = (_Float16*)d_ws;   // 25.6 MB f16 gather table

  int b1 = (total + 255) / 256;
  a2_normalize<<<b1, 256, 0, stream>>>(feat, out, h16, total);

  int b2 = (num_node + WPB - 1) / WPB;
  a2_route<<<b2, 256, 0, stream>>>(out, h16, neighbors, niter, out, num_node);
}
